// RelativeSelfAttention_85718957293717
// MI455X (gfx1250) — matrix operand order repaired and checked by reading, not compiled, not run on hardware
//
#include <hip/hip_runtime.h>
#include <math.h>

// Problem constants (from reference): T=2048, B=2, D=1024, H=16, DH=64
constexpr int Tn  = 2048;
constexpr int Bn  = 2;
constexpr int Dn  = 1024;
constexpr int Hn  = 16;
constexpr int DHn = 64;

typedef __attribute__((ext_vector_type(16))) _Float16 v16h;
typedef __attribute__((ext_vector_type(8)))  float    v8f;
typedef __attribute__((ext_vector_type(4)))  unsigned int u32x4;
typedef __attribute__((ext_vector_type(8)))  unsigned int u32x8;

// ---------------------------------------------------------------------------
// LDS byte-offset of a generic pointer to a __shared__ object (AS3 ptrs are
// 32-bit offsets from the workgroup LDS base on AMDGPU).
// ---------------------------------------------------------------------------
__device__ __forceinline__ unsigned lds_off(const void* p) {
  return (unsigned)(unsigned long long)(const __attribute__((address_space(3))) void*)p;
}

// ---------------------------------------------------------------------------
// WMMA fragment helpers (wave32, v_wmma_f32_16x16x32_f16 layouts per ISA 7.12.2)
// ---------------------------------------------------------------------------

union FragU { v16h v; unsigned int u[8]; _Float16 h[16]; };

// A: 16x32 (MxK) tile from row-major f16 source, ld = row stride (elements).
__device__ __forceinline__ v16h load_frag_a_f16(const _Float16* __restrict__ src, int ld) {
  const int lane = threadIdx.x & 31;
  const int m    = lane & 15;
  const int hi   = lane >> 4;
  FragU f;
#pragma unroll
  for (int i = 0; i < 8; ++i) {
    const int k = ((i >= 4) ? 16 : 0) + (hi ? 8 : 0) + 2 * (i & 3);
    f.u[i] = *reinterpret_cast<const unsigned int*>(src + (size_t)m * ld + k);
  }
  return f.v;
}

// A tile from row-major f32 source, converted to f16 in registers.
__device__ __forceinline__ v16h load_frag_a_f32(const float* __restrict__ src, int ld) {
  const int lane = threadIdx.x & 31;
  const int m    = lane & 15;
  const int hi   = lane >> 4;
  FragU f;
#pragma unroll
  for (int i = 0; i < 8; ++i) {
    const int k = ((i >= 4) ? 16 : 0) + (hi ? 8 : 0) + 2 * (i & 3);
    const float2 p = *reinterpret_cast<const float2*>(src + (size_t)m * ld + k);
    f.h[2 * i]     = (_Float16)p.x;
    f.h[2 * i + 1] = (_Float16)p.y;
  }
  return f.v;
}

// B: 32x16 (KxN) tile where memory holds Bt[N][K] row-major (C = A * Bt^T).
__device__ __forceinline__ v16h load_frag_b_nk(const _Float16* src, int ld) {
  const int lane = threadIdx.x & 31;
  const int n    = lane & 15;
  const int hi   = lane >> 4;
  FragU f;
#pragma unroll
  for (int i = 0; i < 8; ++i) {
    const int k = ((i >= 4) ? 16 : 0) + (hi ? 8 : 0) + 2 * (i & 3);
    f.u[i] = *reinterpret_cast<const unsigned int*>(src + (size_t)n * ld + k);
  }
  return f.v;
}

// Same, with global row (n0 + n) clamped to nmax (OOB rows only feed masked outputs).
__device__ __forceinline__ v16h load_frag_b_nk_clamp(const _Float16* __restrict__ base,
                                                     int n0, int ld, int nmax) {
  const int lane = threadIdx.x & 31;
  int n = n0 + (lane & 15);
  if (n > nmax) n = nmax;
  const int hi = lane >> 4;
  FragU f;
#pragma unroll
  for (int i = 0; i < 8; ++i) {
    const int k = ((i >= 4) ? 16 : 0) + (hi ? 8 : 0) + 2 * (i & 3);
    f.u[i] = *reinterpret_cast<const unsigned int*>(base + (size_t)n * ld + k);
  }
  return f.v;
}

__device__ __forceinline__ v8f wmma_f16(v16h a, v16h b, v8f c) {
  return __builtin_amdgcn_wmma_f32_16x16x32_f16(false, a, false, b, (short)0, c,
                                                false, false);
}

// ---------------------------------------------------------------------------
// TDM: DMA a [rows=64 x k=32] f16 panel (row stride Dn) from global to LDS.
// D# per ISA 8.3/8.4: group0 {count=1, lds_addr, global_addr, type=2},
// group1 {data_size=2B, tensor_dim0/1, tile_dim0=32, tile_dim1=64, stride0=Dn}.
// ---------------------------------------------------------------------------
__device__ __forceinline__ void tdm_load_panel(const _Float16* gsrc, unsigned ldsByteOff) {
  const unsigned long long ga = (unsigned long long)(const void*)gsrc;
  u32x4 g0;
  g0[0] = 0x1u;                                           // count=1, user mode
  g0[1] = ldsByteOff;                                     // lds_addr (bytes)
  g0[2] = (unsigned)ga;                                   // global_addr[31:0]
  g0[3] = ((unsigned)(ga >> 32) & 0x01FFFFFFu) | 0x80000000u;  // addr[56:32] | type=2
  u32x8 g1;
  g1[0] = (1u << 16);                                     // data_size = 1 -> 2 bytes
  g1[1] = ((unsigned)Dn & 0xFFFFu) << 16;                 // tensor_dim0[15:0]
  g1[2] = (((unsigned)Dn >> 16) & 0xFFFFu) | (4096u << 16); // td0 hi | tensor_dim1 lo
  g1[3] = 0u | (32u << 16);                               // td1 hi | tile_dim0 = 32
  g1[4] = 64u;                                            // tile_dim1 = 64, tile_dim2 = 0
  g1[5] = (unsigned)Dn;                                   // tensor_dim0_stride lo
  g1[6] = 0u;                                             // stride0 hi | stride1 lo
  g1[7] = 0u;
  asm volatile("tensor_load_to_lds %0, %1" :: "s"(g0), "s"(g1) : "memory");
}

// ---------------------------------------------------------------------------
// Elementwise: f32 -> f16 cast (weights, pos)
// ---------------------------------------------------------------------------
__global__ void cast_kernel(const float* __restrict__ s, _Float16* __restrict__ d, int n) {
  int i = blockIdx.x * 256 + threadIdx.x;
  const int stride = gridDim.x * 256;
  for (; i < n; i += stride) d[i] = (_Float16)s[i];
}

// ---------------------------------------------------------------------------
// LayerNorm over D=1024, output f16.  One 256-thread block per row (t*B+b).
// ---------------------------------------------------------------------------
__global__ void ln_cast_kernel(const float* __restrict__ x, const float* __restrict__ g,
                               const float* __restrict__ beta, _Float16* __restrict__ xh) {
  const int row = blockIdx.x;
  const int tid = threadIdx.x;
  const float* xr = x + (size_t)row * Dn;
  float vals[4];
  float s = 0.f, ss = 0.f;
#pragma unroll
  for (int i = 0; i < 4; ++i) {
    const float v = xr[tid + 256 * i];
    vals[i] = v; s += v; ss += v * v;
  }
  __shared__ float rs[256], rss[256];
  rs[tid] = s; rss[tid] = ss;
  __syncthreads();
  for (int o = 128; o > 0; o >>= 1) {
    if (tid < o) { rs[tid] += rs[tid + o]; rss[tid] += rss[tid + o]; }
    __syncthreads();
  }
  const float mu  = rs[0] * (1.f / Dn);
  const float var = rss[0] * (1.f / Dn) - mu * mu;
  const float inv = rsqrtf(var + 1e-5f);
#pragma unroll
  for (int i = 0; i < 4; ++i) {
    const int c = tid + 256 * i;
    xh[(size_t)row * Dn + c] = (_Float16)(((vals[i] - mu) * inv) * g[c] + beta[c]);
  }
}

// ---------------------------------------------------------------------------
// QKV GEMM: [4096,1024]f16 @ [3072,1024]^T -> scatter epilogue.
// One wave computes a 16x64 output strip (4 accumulators, A reused).
// ---------------------------------------------------------------------------
__global__ void gemm_qkv_kernel(const _Float16* __restrict__ Xh, const _Float16* __restrict__ Wh,
                                const float* __restrict__ bias,
                                const float* __restrict__ rwb, const float* __restrict__ rrb,
                                _Float16* __restrict__ qw, _Float16* __restrict__ qr,
                                _Float16* __restrict__ kk, _Float16* __restrict__ vT) {
  const int gw = (blockIdx.x * blockDim.x + threadIdx.x) >> 5;
  constexpr int NS = (3 * Dn) / 64;              // 48 column strips
  const int m0 = (gw / NS) * 16;
  const int n0 = (gw % NS) * 64;
  v8f acc[4] = {};
  const _Float16* Arow = Xh + (size_t)m0 * Dn;
  for (int k = 0; k < Dn; k += 32) {
    __builtin_prefetch(Arow + k + 64, 0, 0);     // global_prefetch_b8
    const v16h a = load_frag_a_f16(Arow + k, Dn);
#pragma unroll
    for (int s = 0; s < 4; ++s) {
      const v16h b = load_frag_b_nk(Wh + (size_t)(n0 + 16 * s) * Dn + k, Dn);
      acc[s] = wmma_f16(a, b, acc[s]);
    }
  }
  const int lane = threadIdx.x & 31;
  const int nloc = lane & 15;
  const int mofs = (lane >> 4) ? 8 : 0;
#pragma unroll
  for (int s = 0; s < 4; ++s) {
#pragma unroll
    for (int i = 0; i < 8; ++i) {
      const int m_g = m0 + i + mofs;
      const int n_g = n0 + 16 * s + nloc;
      float v = acc[s][i] + bias[n_g];
      const int t = m_g / Bn, b = m_g % Bn;
      const int sel = n_g / Dn;
      const int inner = n_g % Dn;
      const int h = inner / DHn, dh = inner % DHn;
      const size_t bh = (size_t)(b * Hn + h);
      if (sel == 0) {
        qw[(bh * Tn + t) * DHn + dh] = (_Float16)(v + rwb[inner]);
        qr[(bh * Tn + t) * DHn + dh] = (_Float16)(v + rrb[inner]);
      } else if (sel == 1) {
        kk[(bh * Tn + t) * DHn + dh] = (_Float16)v;
      } else {
        vT[(bh * DHn + dh) * Tn + t] = (_Float16)v;
      }
    }
  }
}

// ---------------------------------------------------------------------------
// Pos GEMM: r = pos @ pos_proj_w^T, stored as rh[H][T][64] f16.
// ---------------------------------------------------------------------------
__global__ void gemm_pos_kernel(const _Float16* __restrict__ Ph, const _Float16* __restrict__ Wp,
                                _Float16* __restrict__ rh) {
  const int gw = (blockIdx.x * blockDim.x + threadIdx.x) >> 5;
  constexpr int NS = Dn / 64;                    // 16 strips
  const int m0 = (gw / NS) * 16;                 // t tile
  const int n0 = (gw % NS) * 64;
  v8f acc[4] = {};
  const _Float16* Arow = Ph + (size_t)m0 * Dn;
  for (int k = 0; k < Dn; k += 32) {
    const v16h a = load_frag_a_f16(Arow + k, Dn);
#pragma unroll
    for (int s = 0; s < 4; ++s) {
      const v16h b = load_frag_b_nk(Wp + (size_t)(n0 + 16 * s) * Dn + k, Dn);
      acc[s] = wmma_f16(a, b, acc[s]);
    }
  }
  const int lane = threadIdx.x & 31;
  const int nloc = lane & 15;
  const int mofs = (lane >> 4) ? 8 : 0;
#pragma unroll
  for (int s = 0; s < 4; ++s) {
#pragma unroll
    for (int i = 0; i < 8; ++i) {
      const int t = m0 + i + mofs;
      const int n_g = n0 + 16 * s + nloc;
      const int h = n_g / DHn, dh = n_g % DHn;
      rh[((size_t)h * Tn + t) * DHn + dh] = (_Float16)acc[s][i];
    }
  }
}

// ---------------------------------------------------------------------------
// Score tiles: ac = qw·k^T; bd via rel-shift gather of raw = qr·r^T.
// For j<=i: bd[i,j] = raw[i, j + T-1-i].  A 16x16 tile needs raw columns in
// [cbase, cbase+32) with cbase = T-16-(i0-j0); two raw tiles by WMMA, staged
// in LDS, gathered as lds[m][15+n-m].  One wave per 16x16 tile.
// ---------------------------------------------------------------------------
__global__ void scores_kernel(const _Float16* __restrict__ qw, const _Float16* __restrict__ qr,
                              const _Float16* __restrict__ kk, const _Float16* __restrict__ rh,
                              float* __restrict__ attn) {
  const int jt = blockIdx.x, it = blockIdx.y, bh = blockIdx.z;
  const int i0 = it * 16, j0 = jt * 16;
  const int lane = threadIdx.x & 31;
  const int nloc = lane & 15;
  const int mofs = (lane >> 4) ? 8 : 0;
  float* out = attn + ((size_t)bh * Tn + i0) * Tn + j0;

  if (jt > it) {                                  // tile fully in causal mask
#pragma unroll
    for (int i = 0; i < 8; ++i) out[(size_t)(i + mofs) * Tn + nloc] = -1e9f;
    return;
  }

  const int h = bh % Hn;
  const _Float16* qwb = qw + (size_t)bh * Tn * DHn;
  const _Float16* qrb = qr + (size_t)bh * Tn * DHn;
  const _Float16* kb  = kk + (size_t)bh * Tn * DHn;
  const _Float16* rb  = rh + (size_t)h  * Tn * DHn;

  v8f ac = {}, rA = {}, rB = {};
  const int cbase = Tn - 16 - (i0 - j0);          // >= 0 for all computed tiles
#pragma unroll
  for (int k = 0; k < DHn; k += 32) {
    const v16h aW = load_frag_a_f16(qwb + (size_t)i0 * DHn + k, DHn);
    const v16h bK = load_frag_b_nk(kb + (size_t)j0 * DHn + k, DHn);
    ac = wmma_f16(aW, bK, ac);
    const v16h aR = load_frag_a_f16(qrb + (size_t)i0 * DHn + k, DHn);
    const v16h bA = load_frag_b_nk_clamp(rb + k, cbase,      DHn, Tn - 1);
    rA = wmma_f16(aR, bA, rA);
    const v16h bB = load_frag_b_nk_clamp(rb + k, cbase + 16, DHn, Tn - 1);
    rB = wmma_f16(aR, bB, rB);
  }

  __shared__ float lds[16][32];
#pragma unroll
  for (int i = 0; i < 8; ++i) {
    lds[i + mofs][nloc]      = rA[i];
    lds[i + mofs][nloc + 16] = rB[i];
  }
  __syncthreads();
#pragma unroll
  for (int i = 0; i < 8; ++i) {
    const int m  = i + mofs;
    const int ig = i0 + m, jg = j0 + nloc;
    const float bd = lds[m][15 + nloc - m];
    float sc = (ac[i] + bd) * 0.125f;             // 1/sqrt(64)
    if (jg > ig) sc = -1e9f;
    out[(size_t)m * Tn + nloc] = sc;
  }
}

// ---------------------------------------------------------------------------
// Row softmax over T=2048, in place.  One 256-thread block per row.
// Row is staged global->LDS with GLOBAL_LOAD_ASYNC_TO_LDS_B128 (ASYNCcnt).
// ---------------------------------------------------------------------------
__global__ void softmax_kernel(float* __restrict__ attn) {
  __shared__ alignas(16) float rowbuf[Tn];        // 8 KB
  __shared__ float red[256];
  float* p = attn + (size_t)blockIdx.x * Tn;
  const int tid = threadIdx.x;

  {
    const unsigned l0 = lds_off(&rowbuf[tid * 4]);
    const float*   g0 = p + tid * 4;
    asm volatile("global_load_async_to_lds_b128 %0, %1, off" :: "v"(l0), "v"(g0) : "memory");
    const unsigned l1 = l0 + 4096;
    const float*   g1 = g0 + 1024;
    asm volatile("global_load_async_to_lds_b128 %0, %1, off" :: "v"(l1), "v"(g1) : "memory");
    asm volatile("s_wait_asynccnt 0x0" ::: "memory");
  }
  __syncthreads();

  float v[8];
  float mx = -3.4e38f;
#pragma unroll
  for (int i = 0; i < 8; ++i) { v[i] = rowbuf[tid + 256 * i]; mx = fmaxf(mx, v[i]); }
  red[tid] = mx;
  __syncthreads();
  for (int o = 128; o > 0; o >>= 1) {
    if (tid < o) red[tid] = fmaxf(red[tid], red[tid + o]);
    __syncthreads();
  }
  mx = red[0];
  __syncthreads();
  float s = 0.f;
#pragma unroll
  for (int i = 0; i < 8; ++i) { v[i] = __expf(v[i] - mx); s += v[i]; }
  red[tid] = s;
  __syncthreads();
  for (int o = 128; o > 0; o >>= 1) {
    if (tid < o) red[tid] += red[tid + o];
    __syncthreads();
  }
  const float inv = 1.f / red[0];
#pragma unroll
  for (int i = 0; i < 8; ++i) p[tid + 256 * i] = v[i] * inv;
}

// ---------------------------------------------------------------------------
// ctx = attn @ v.  A = f32 attn rows (converted to f16 fragments), B from
// transposed vT (contiguous K).  Causal truncation of the K loop.
// ---------------------------------------------------------------------------
__global__ void av_kernel(const float* __restrict__ attn, const _Float16* __restrict__ vT,
                          _Float16* __restrict__ ctxh) {
  const int gw = (blockIdx.x * blockDim.x + threadIdx.x) >> 5;
  const int it = gw % (Tn / 16);
  const int bh = gw / (Tn / 16);
  const int i0 = it * 16;
  const float*    Ab = attn + ((size_t)bh * Tn + i0) * Tn;
  const _Float16* Vb = vT + (size_t)bh * DHn * Tn;
  v8f acc[4] = {};
  const int kend = i0 + 16;                       // attn[i, j>i] == 0 after softmax
  for (int k = 0; k < kend; k += 32) {
    const v16h a = load_frag_a_f32(Ab + k, Tn);
#pragma unroll
    for (int s = 0; s < 4; ++s) {
      const v16h b = load_frag_b_nk(Vb + (size_t)(16 * s) * Tn + k, Tn);
      acc[s] = wmma_f16(a, b, acc[s]);
    }
  }
  const int lane = threadIdx.x & 31;
  const int nloc = lane & 15;
  const int mofs = (lane >> 4) ? 8 : 0;
  const int b = bh / Hn, h = bh % Hn;
#pragma unroll
  for (int s = 0; s < 4; ++s) {
#pragma unroll
    for (int i = 0; i < 8; ++i) {
      const int t  = i0 + i + mofs;
      const int dh = 16 * s + nloc;
      ctxh[((size_t)t * Bn + b) * Dn + h * DHn + dh] = (_Float16)acc[s][i];
    }
  }
}

// ---------------------------------------------------------------------------
// Output GEMM: out = ctx @ out_proj_w^T + b, f32 result.
// Block = 8 waves sharing one 64-col weight strip; wave 0 stages the weight
// panel per 32-deep K chunk into double-buffered LDS via the Tensor Data
// Mover; all waves read B fragments from LDS (ds loads).
// ---------------------------------------------------------------------------
__global__ void gemm_out_kernel(const _Float16* __restrict__ ctxh, const _Float16* __restrict__ Wo,
                                const float* __restrict__ bias, float* __restrict__ out) {
  __shared__ alignas(16) _Float16 panel[2][64 * 32];   // 2 x 4 KB
  const int ns = blockIdx.x & 15;
  const int mb = blockIdx.x >> 4;
  const int w  = threadIdx.x >> 5;
  const int m0 = (mb * 8 + w) * 16;
  const int n0 = ns * 64;
  const bool lead = (w == 0);

  if (lead) tdm_load_panel(Wo + (size_t)n0 * Dn, lds_off(&panel[0][0]));

  v8f acc[4] = {};
  const _Float16* Arow = ctxh + (size_t)m0 * Dn;
  int cur = 0;
  for (int k = 0; k < Dn; k += 32) {
    if (lead) __builtin_amdgcn_s_wait_tensorcnt(0);
    __syncthreads();                               // panel[cur] ready for all waves
    if (lead && (k + 32) < Dn)
      tdm_load_panel(Wo + (size_t)n0 * Dn + (k + 32), lds_off(&panel[cur ^ 1][0]));
    const v16h a = load_frag_a_f16(Arow + k, Dn);
#pragma unroll
    for (int s = 0; s < 4; ++s) {
      const v16h b = load_frag_b_nk(&panel[cur][(16 * s) * 32], 32);
      acc[s] = wmma_f16(a, b, acc[s]);
    }
    cur ^= 1;
  }

  const int lane = threadIdx.x & 31;
  const int nloc = lane & 15;
  const int mofs = (lane >> 4) ? 8 : 0;
#pragma unroll
  for (int s = 0; s < 4; ++s) {
#pragma unroll
    for (int i = 0; i < 8; ++i) {
      const int m_g = m0 + i + mofs;
      const int n_g = n0 + 16 * s + nloc;
      out[(size_t)m_g * Dn + n_g] = acc[s][i] + bias[n_g];
    }
  }
}

// ---------------------------------------------------------------------------
// Host launch
// ---------------------------------------------------------------------------
extern "C" void kernel_launch(void* const* d_in, const int* in_sizes, int n_in,
                              void* d_out, int out_size, void* d_ws, size_t ws_size,
                              hipStream_t stream) {
  (void)in_sizes; (void)n_in; (void)out_size; (void)ws_size;
  const float* input = (const float*)d_in[0];
  const float* pos   = (const float*)d_in[1];
  const float* ln_g  = (const float*)d_in[2];
  const float* ln_b  = (const float*)d_in[3];
  const float* w_in  = (const float*)d_in[4];
  const float* b_in  = (const float*)d_in[5];
  const float* w_pos = (const float*)d_in[6];
  const float* rwb   = (const float*)d_in[7];
  const float* rrb   = (const float*)d_in[8];
  const float* w_out = (const float*)d_in[9];
  const float* b_out = (const float*)d_in[10];
  // d_in[11] = attn_mask (causal), computed analytically in-kernel.

  char* ws = (char*)d_ws;
  size_t off = 0;
  auto alloc = [&](size_t bytes) -> void* {
    void* p = ws + off;
    off += (bytes + 255) & ~(size_t)255;
    return p;
  };
  _Float16* Whin  = (_Float16*)alloc((size_t)3 * Dn * Dn * 2);
  _Float16* Whpos = (_Float16*)alloc((size_t)Dn * Dn * 2);
  _Float16* Whout = (_Float16*)alloc((size_t)Dn * Dn * 2);
  _Float16* Xh    = (_Float16*)alloc((size_t)Tn * Bn * Dn * 2);
  _Float16* Ph    = (_Float16*)alloc((size_t)Tn * Dn * 2);
  _Float16* qw    = (_Float16*)alloc((size_t)Bn * Hn * Tn * DHn * 2);
  _Float16* qr    = (_Float16*)alloc((size_t)Bn * Hn * Tn * DHn * 2);
  _Float16* kkb   = (_Float16*)alloc((size_t)Bn * Hn * Tn * DHn * 2);
  _Float16* vT    = (_Float16*)alloc((size_t)Bn * Hn * DHn * Tn * 2);
  _Float16* rh    = (_Float16*)alloc((size_t)Hn * Tn * DHn * 2);
  _Float16* ctxh  = (_Float16*)alloc((size_t)Tn * Bn * Dn * 2);

  float* out_p  = (float*)d_out;                       // [T,B,D]
  float* attn_p = out_p + (size_t)Tn * Bn * Dn;        // [B,H,T,T]

  cast_kernel<<<512, 256, 0, stream>>>(w_in,  Whin,  3 * Dn * Dn);
  cast_kernel<<<256, 256, 0, stream>>>(w_pos, Whpos, Dn * Dn);
  cast_kernel<<<256, 256, 0, stream>>>(w_out, Whout, Dn * Dn);
  cast_kernel<<<512, 256, 0, stream>>>(pos,   Ph,    Tn * Dn);

  ln_cast_kernel<<<Tn * Bn, 256, 0, stream>>>(input, ln_g, ln_b, Xh);

  // 256 m-tiles x 48 n-strips = 12288 waves, 8 waves/block
  gemm_qkv_kernel<<<(256 * 48) / 8, 256, 0, stream>>>(Xh, Whin, b_in, rwb, rrb,
                                                      qw, qr, kkb, vT);
  // 128 m-tiles x 16 n-strips
  gemm_pos_kernel<<<(128 * 16) / 8, 256, 0, stream>>>(Ph, Whpos, rh);

  scores_kernel<<<dim3(Tn / 16, Tn / 16, Bn * Hn), 32, 0, stream>>>(qw, qr, kkb, rh, attn_p);

  softmax_kernel<<<Bn * Hn * Tn, 256, 0, stream>>>(attn_p);

  av_kernel<<<(Bn * Hn * (Tn / 16)) / 8, 256, 0, stream>>>(attn_p, vT, ctxh);

  // 32 m-blocks (8 waves each) x 16 n-strips
  gemm_out_kernel<<<512, 256, 0, stream>>>(ctxh, Whout, b_out, out_p);
}
